// CompositionalEmbedder_51462298140786
// MI455X (gfx1250) — matrix operand order repaired and checked by reading, not compiled, not run on hardware
//
#include <hip/hip_runtime.h>
#include <hip/hip_bf16.h>

typedef __bf16 bf16_t;
typedef __attribute__((ext_vector_type(16))) __bf16 v16bf;
typedef __attribute__((ext_vector_type(8)))  float  v8f;

// Problem constants (from reference)
#define BATCH     8
#define S_LEN     4096
#define I_LEN     256
#define STEP      4
#define D_MODEL   2048
#define NG        960          // groups per batch
#define NTOK      1216         // I_LEN + NG
#define NGROUPS   7680         // BATCH * NG
#define MROWS_KV  30720        // NGROUPS * STEP
#define HEADS     64
#define HDIM      32

__device__ __forceinline__ bf16_t f2bf(float f) {
    unsigned u = __float_as_uint(f);
    unsigned r = (u + 0x7FFFu + ((u >> 16) & 1u)) >> 16;
    unsigned short s = (unsigned short)r;
    return __builtin_bit_cast(bf16_t, s);
}
__device__ __forceinline__ float bf2f(bf16_t b) {
    unsigned short s = __builtin_bit_cast(unsigned short, b);
    return __uint_as_float(((unsigned)s) << 16);
}

// CDNA5 async global->LDS copy (16 B per lane), tracked by ASYNCcnt.
__device__ __forceinline__ void async_cp16(void* lds, const void* g) {
    unsigned l = (unsigned)(uintptr_t)lds;                 // LDS byte offset
    unsigned long long a = (unsigned long long)(uintptr_t)g;
    asm volatile("global_load_async_to_lds_b128 %0, %1, off"
                 :: "v"(l), "v"(a) : "memory");
}
__device__ __forceinline__ void wait_async0() {
    asm volatile("s_wait_asynccnt 0x0" ::: "memory");
}

// ---------------------------------------------------------------------------
// Weight transpose + f32->bf16 convert:  Wt[n][k] = W[k][n], 2048x2048
// ---------------------------------------------------------------------------
__global__ void transpose_bf16_kernel(const float* __restrict__ W,
                                      bf16_t* __restrict__ Wt) {
    int idx = blockIdx.x * blockDim.x + threadIdx.x;   // over 2048*2048
    int k = idx & (D_MODEL - 1);
    int n = idx >> 11;
    Wt[idx] = f2bf(W[(size_t)k * D_MODEL + n]);
}

// ---------------------------------------------------------------------------
// Fold pos_emb @ Wk[2048:,:] (+bk) and pos_emb @ Wv[2048:,:] (+bv) into
// per-g bias tables of shape (4, 2048).
// ---------------------------------------------------------------------------
__global__ void pe_bias_kernel(const float* __restrict__ pos_emb,
                               const float* __restrict__ Wk, const float* __restrict__ bk,
                               const float* __restrict__ Wv, const float* __restrict__ bv,
                               float* __restrict__ pkb, float* __restrict__ pvb) {
    int idx = blockIdx.x * blockDim.x + threadIdx.x;   // over 4*2048
    int n = idx & (D_MODEL - 1);
    int g = idx >> 11;
    float sk = bk[n], sv = bv[n];
    for (int j = 0; j < D_MODEL / 2; ++j) {
        float p = pos_emb[g * (D_MODEL / 2) + j];
        sk += p * Wk[(size_t)(D_MODEL + j) * D_MODEL + n];
        sv += p * Wv[(size_t)(D_MODEL + j) * D_MODEL + n];
    }
    pkb[idx] = sk;
    pvb[idx] = sv;
}

// ---------------------------------------------------------------------------
// Embedding gather + group mean; emits f32 mean (residual), bf16 mean (GEMM A)
// and bf16 group tokens (K/V GEMM A).
// ---------------------------------------------------------------------------
__global__ void gather_mean_kernel(const int* __restrict__ ids,
                                   const float* __restrict__ emb,
                                   float* __restrict__ mean_f32,
                                   bf16_t* __restrict__ mean_bf,
                                   bf16_t* __restrict__ groups_bf) {
    int idx = blockIdx.x * blockDim.x + threadIdx.x;   // over NGROUPS * D_MODEL
    int d = idx & (D_MODEL - 1);
    int n = idx >> 11;
    int b = n / NG;
    int i = n - b * NG;
    int base = b * S_LEN + I_LEN + i * STEP;
    float s = 0.f;
#pragma unroll
    for (int g = 0; g < STEP; ++g) {
        int row = ids[base + g];
        float v = emb[(size_t)row * D_MODEL + d];
        s += v;
        groups_bf[((size_t)(n * STEP + g)) * D_MODEL + d] = f2bf(v);
    }
    s *= 0.25f;
    mean_f32[idx] = s;
    mean_bf[idx]  = f2bf(s);
}

// ---------------------------------------------------------------------------
// Instruction-token passthrough: out[b, t, :] = emb_table[ids[b*S + t]]
// ---------------------------------------------------------------------------
__global__ void inst_copy_kernel(const int* __restrict__ ids,
                                 const float* __restrict__ emb,
                                 float* __restrict__ out) {
    int idx = blockIdx.x * blockDim.x + threadIdx.x;   // over BATCH*I_LEN*D_MODEL
    int d = idx & (D_MODEL - 1);
    int r = idx >> 11;
    int b = r >> 8;
    int t = r & (I_LEN - 1);
    int row = ids[b * S_LEN + t];
    out[((size_t)(b * NTOK + t)) * D_MODEL + d] = emb[(size_t)row * D_MODEL + d];
}

// ---------------------------------------------------------------------------
// bf16 WMMA GEMM:  C(MxN) = A(MxK) * Bt(NxK)^T   with K = N = 2048
// Block tile 128x256x32; 256 threads = 8 waves (2 in M x 4 in N);
// each wave: 4x4 tiles of v_wmma_f32_16x16x32_bf16 (64x64 wave tile).
// Double-buffered LDS filled with global_load_async_to_lds_b128 (ASYNCcnt).
// MODE 0: += biasVec[n],                   store bf16
// MODE 1: += biasMat[(m&3)*2048+n],        store bf16
// MODE 2: += biasVec[n] + resid[m][n],     store f32 scattered to output rows
// ---------------------------------------------------------------------------
#define BM 128
#define BN 256
#define BK 32
#define LDP (BK + 8)   // padded LDS pitch: 40 elems = 80 B, conflict-free

__global__ __launch_bounds__(256)
void gemm_bf16_kernel(const bf16_t* __restrict__ A,
                      const bf16_t* __restrict__ Bt,
                      const float* __restrict__ biasVec,
                      const float* __restrict__ biasMat,
                      const float* __restrict__ resid,
                      bf16_t* __restrict__ outB,
                      float* __restrict__ outF,
                      int MODE) {
    __shared__ bf16_t As[2][BM][LDP];   // 2 * 128*40*2 = 20.0 KB
    __shared__ bf16_t Bs[2][BN][LDP];   // 2 * 256*40*2 = 40.0 KB

    const int t    = threadIdx.x;
    const int lane = t & 31;
    const int w    = t >> 5;
    const int wm   = w & 1;        // wave row (0..1)  -> 64 rows each
    const int wn   = w >> 1;       // wave col (0..3)  -> 64 cols each
    const int m0   = blockIdx.y * BM;
    const int n0   = blockIdx.x * BN;

    const bf16_t* gA = A  + (size_t)m0 * D_MODEL;
    const bf16_t* gB = Bt + (size_t)n0 * D_MODEL;

    v8f acc[4][4];
#pragma unroll
    for (int i = 0; i < 4; ++i)
#pragma unroll
        for (int j = 0; j < 4; ++j)
            acc[i][j] = (v8f)0.0f;

    // ---- async stage of K-tile k0 into buffer buf ----
    auto stage = [&](int buf, int k0) {
        // A tile: 128 rows x 32 cols = 512 x 16B chunks, 2 per thread
#pragma unroll
        for (int c = 0; c < 2; ++c) {
            int ch = t + c * 256;
            int r  = ch >> 2;
            int o  = (ch & 3) * 8;
            async_cp16(&As[buf][r][o], gA + (size_t)r * D_MODEL + k0 + o);
        }
        // B tile: 256 rows x 32 cols = 1024 x 16B chunks, 4 per thread
#pragma unroll
        for (int c = 0; c < 4; ++c) {
            int ch = t + c * 256;
            int r  = ch >> 2;
            int o  = (ch & 3) * 8;
            async_cp16(&Bs[buf][r][o], gB + (size_t)r * D_MODEL + k0 + o);
        }
    };

    stage(0, 0);
    wait_async0();
    __syncthreads();

    for (int k0 = 0; k0 < D_MODEL; k0 += BK) {
        const int cur = (k0 / BK) & 1;
        if (k0 + BK < D_MODEL)
            stage(cur ^ 1, k0 + BK);   // overlap DMA of next tile with WMMA

        // A fragments: lane<16 -> K {0..7,16..23}; lane>=16 -> K {8..15,24..31}
        v16bf af[4];
#pragma unroll
        for (int tm = 0; tm < 4; ++tm) {
            int r  = wm * 64 + tm * 16 + (lane & 15);
            int kc = (lane >> 4) * 8;
            *((uint4*)&af[tm])     = *(const uint4*)&As[cur][r][kc];
            *((uint4*)&af[tm] + 1) = *(const uint4*)&As[cur][r][kc + 16];
        }
        // B fragments: lane holds column n = lane&15; lane<16 -> K {0..7,16..23}; lane>=16 -> K {8..15,24..31}
        v16bf bfr[4];
#pragma unroll
        for (int tn = 0; tn < 4; ++tn) {
            int c  = wn * 64 + tn * 16 + (lane & 15);
            int kb = (lane >> 4) * 8;
            *((uint4*)&bfr[tn])     = *(const uint4*)&Bs[cur][c][kb];
            *((uint4*)&bfr[tn] + 1) = *(const uint4*)&Bs[cur][c][kb + 16];
        }
#pragma unroll
        for (int tm = 0; tm < 4; ++tm)
#pragma unroll
            for (int tn = 0; tn < 4; ++tn)
                acc[tm][tn] = __builtin_amdgcn_wmma_f32_16x16x32_bf16(
                    false, af[tm], false, bfr[tn], (short)0, acc[tm][tn],
                    false, false);

        wait_async0();      // next tile fully landed in LDS
        __syncthreads();    // all waves done reading current tile
    }

    // Epilogue. C 16x16 f32 layout: VGPR r, lane L -> (m = r + 8*(L/16), n = L%16)
    const int lm = (lane >> 4) * 8;
    const int ln = lane & 15;
#pragma unroll
    for (int tm = 0; tm < 4; ++tm) {
#pragma unroll
        for (int tn = 0; tn < 4; ++tn) {
#pragma unroll
            for (int r = 0; r < 8; ++r) {
                int mm = m0 + wm * 64 + tm * 16 + lm + r;
                int nn = n0 + wn * 64 + tn * 16 + ln;
                float cv = acc[tm][tn][r];
                if (MODE == 0) {
                    cv += biasVec[nn];
                    outB[(size_t)mm * D_MODEL + nn] = f2bf(cv);
                } else if (MODE == 1) {
                    cv += biasMat[(mm & 3) * D_MODEL + nn];
                    outB[(size_t)mm * D_MODEL + nn] = f2bf(cv);
                } else {
                    cv += biasVec[nn] + resid[(size_t)mm * D_MODEL + nn];
                    int bb = mm / NG;
                    int ii = mm - bb * NG;
                    size_t orow = (size_t)bb * NTOK + I_LEN + ii;
                    outF[orow * D_MODEL + nn] = cv;
                }
            }
        }
    }
}

// ---------------------------------------------------------------------------
// Attention: one wave per (n, h); lane = d (0..31). 4-way softmax over groups.
// ---------------------------------------------------------------------------
__global__ void attn_kernel(const bf16_t* __restrict__ q,
                            const bf16_t* __restrict__ k,
                            const bf16_t* __restrict__ v,
                            bf16_t* __restrict__ outb) {
    int wave = blockIdx.x * (blockDim.x >> 5) + (threadIdx.x >> 5);
    int lane = threadIdx.x & 31;
    int h = wave & (HEADS - 1);
    int n = wave >> 6;
    int col = h * HDIM + lane;

    float qv = bf2f(q[(size_t)n * D_MODEL + col]);
    float sc[STEP];
#pragma unroll
    for (int g = 0; g < STEP; ++g) {
        float kv = bf2f(k[((size_t)(n * STEP + g)) * D_MODEL + col]);
        float p = qv * kv;
#pragma unroll
        for (int off = 16; off > 0; off >>= 1)
            p += __shfl_xor(p, off, 32);
        sc[g] = p * 0.17677669529663687f;   // 1/sqrt(32)
    }
    float mx = sc[0];
#pragma unroll
    for (int g = 1; g < STEP; ++g) mx = fmaxf(mx, sc[g]);
    float e[STEP], sum = 0.f;
#pragma unroll
    for (int g = 0; g < STEP; ++g) { e[g] = __expf(sc[g] - mx); sum += e[g]; }
    float inv = 1.0f / sum;
    float o = 0.f;
#pragma unroll
    for (int g = 0; g < STEP; ++g)
        o += e[g] * inv * bf2f(v[((size_t)(n * STEP + g)) * D_MODEL + col]);
    outb[(size_t)n * D_MODEL + col] = f2bf(o);
}

// ---------------------------------------------------------------------------
// position_ids (written as float into the tail of d_out)
// ---------------------------------------------------------------------------
__global__ void posids_kernel(float* __restrict__ outPos) {
    int idx = blockIdx.x * blockDim.x + threadIdx.x;   // over BATCH*NTOK
    int j = idx % NTOK;
    int val = (j < I_LEN) ? j : (I_LEN - 1) + STEP * (j - I_LEN + 1);
    outPos[idx] = (float)val;
}

// ---------------------------------------------------------------------------
extern "C" void kernel_launch(void* const* d_in, const int* in_sizes, int n_in,
                              void* d_out, int out_size, void* d_ws, size_t ws_size,
                              hipStream_t stream) {
    (void)in_sizes; (void)n_in; (void)out_size; (void)ws_size;

    const int*   ids     = (const int*)  d_in[0];
    const float* emb     = (const float*)d_in[4];
    const float* Wq      = (const float*)d_in[5];
    const float* bq      = (const float*)d_in[6];
    const float* Wk      = (const float*)d_in[7];
    const float* bk      = (const float*)d_in[8];
    const float* Wv      = (const float*)d_in[9];
    const float* bv      = (const float*)d_in[10];
    const float* pos_emb = (const float*)d_in[11];
    const float* Wo      = (const float*)d_in[12];
    const float* bo      = (const float*)d_in[13];
    float* out = (float*)d_out;

    // workspace layout
    size_t off = 0;
    auto wsalloc = [&](size_t bytes) {
        size_t o = off;
        off += (bytes + 255) & ~(size_t)255;
        return (char*)d_ws + o;
    };
    float*  mean_f32  = (float*) wsalloc((size_t)NGROUPS * D_MODEL * 4);
    bf16_t* mean_bf   = (bf16_t*)wsalloc((size_t)NGROUPS * D_MODEL * 2);
    bf16_t* groups_bf = (bf16_t*)wsalloc((size_t)MROWS_KV * D_MODEL * 2);
    bf16_t* WtQ       = (bf16_t*)wsalloc((size_t)D_MODEL * D_MODEL * 2);
    bf16_t* WtK       = (bf16_t*)wsalloc((size_t)D_MODEL * D_MODEL * 2);
    bf16_t* WtV       = (bf16_t*)wsalloc((size_t)D_MODEL * D_MODEL * 2);
    bf16_t* WtO       = (bf16_t*)wsalloc((size_t)D_MODEL * D_MODEL * 2);
    float*  pkb       = (float*) wsalloc((size_t)STEP * D_MODEL * 4);
    float*  pvb       = (float*) wsalloc((size_t)STEP * D_MODEL * 4);
    bf16_t* q_bf      = (bf16_t*)wsalloc((size_t)NGROUPS * D_MODEL * 2);
    bf16_t* k_bf      = (bf16_t*)wsalloc((size_t)MROWS_KV * D_MODEL * 2);
    bf16_t* v_bf      = (bf16_t*)wsalloc((size_t)MROWS_KV * D_MODEL * 2);
    bf16_t* o_bf      = (bf16_t*)wsalloc((size_t)NGROUPS * D_MODEL * 2);

    const int TPB = 256;

    // 1) weight transpose+convert
    int wElems = D_MODEL * D_MODEL;
    transpose_bf16_kernel<<<wElems / TPB, TPB, 0, stream>>>(Wq, WtQ);
    transpose_bf16_kernel<<<wElems / TPB, TPB, 0, stream>>>(Wk, WtK);  // top 2048 rows
    transpose_bf16_kernel<<<wElems / TPB, TPB, 0, stream>>>(Wv, WtV);
    transpose_bf16_kernel<<<wElems / TPB, TPB, 0, stream>>>(Wo, WtO);

    // 2) fold pos_emb into K/V biases
    pe_bias_kernel<<<(STEP * D_MODEL) / TPB, TPB, 0, stream>>>(
        pos_emb, Wk, bk, Wv, bv, pkb, pvb);

    // 3) gather + mean, 4) instruction copy
    gather_mean_kernel<<<(NGROUPS * D_MODEL) / TPB, TPB, 0, stream>>>(
        ids, emb, mean_f32, mean_bf, groups_bf);
    inst_copy_kernel<<<(BATCH * I_LEN * D_MODEL) / TPB, TPB, 0, stream>>>(
        ids, emb, out);

    // 5) Q = mean @ Wq + bq
    dim3 gQ(D_MODEL / BN, NGROUPS / BM);       // (8, 60)
    gemm_bf16_kernel<<<gQ, TPB, 0, stream>>>(mean_bf, WtQ, bq, nullptr, nullptr,
                                             q_bf, nullptr, 0);
    // 6) K = groups @ Wk[:2048] + pkb[g]
    dim3 gKV(D_MODEL / BN, MROWS_KV / BM);     // (8, 240)
    gemm_bf16_kernel<<<gKV, TPB, 0, stream>>>(groups_bf, WtK, nullptr, pkb, nullptr,
                                              k_bf, nullptr, 1);
    // 7) V
    gemm_bf16_kernel<<<gKV, TPB, 0, stream>>>(groups_bf, WtV, nullptr, pvb, nullptr,
                                              v_bf, nullptr, 1);

    // 8) attention (wave per (n,h))
    attn_kernel<<<(NGROUPS * HEADS) / 8, TPB, 0, stream>>>(q_bf, k_bf, v_bf, o_bf);

    // 9) merged = mean + out @ Wo + bo, scattered into token_embeds
    gemm_bf16_kernel<<<gQ, TPB, 0, stream>>>(o_bf, WtO, bo, nullptr, mean_f32,
                                             nullptr, out, 2);

    // 10) position ids (float-encoded) after the embeddings
    float* outPos = out + (size_t)BATCH * NTOK * D_MODEL;
    posids_kernel<<<(BATCH * NTOK) / TPB, TPB, 0, stream>>>(outPos);
}
